// ObeliskMIDL_62964220559396
// MI455X (gfx1250) — matrix operand order repaired and checked by reading, not compiled, not run on hardware
//
#include <hip/hip_runtime.h>
#include <hip/hip_bf16.h>

// ---------------------------------------------------------------------------
// ObeliskMIDL forward for MI455X (gfx1250), wave32 + V_WMMA_F32_16X16X32_BF16
// N = 32768 points, channel GEMMs tiled 16(M) x 16(N) x 32(K) per wave.
// ---------------------------------------------------------------------------

#define NPTS   32768
#define NTILES 2048     // NPTS / 16

typedef __attribute__((ext_vector_type(16))) __bf16 v16bf;
typedef __attribute__((ext_vector_type(8)))  float  v8f;

__device__ __forceinline__ unsigned short f2bf(float f) {
  unsigned int u = __float_as_uint(f);
  unsigned int r = u + 0x7fffu + ((u >> 16) & 1u);   // round-to-nearest-even
  return (unsigned short)(r >> 16);
}
__device__ __forceinline__ float bf2f(unsigned short h) {
  return __uint_as_float(((unsigned int)h) << 16);
}

union FragU { uint4 u[2]; v16bf v; };

// Load a 16-element bf16 fragment as two 16B chunks at +0 and +off2 elements.
__device__ __forceinline__ v16bf load_frag(const unsigned short* p, int off2) {
  FragU f;
  f.u[0] = *(const uint4*)(p);
  f.u[1] = *(const uint4*)(p + off2);
  return f.v;
}

// arow pre-offset by m*lda + half*16         -> chunks at +0, +8  (A contiguous)
// brow pre-offset by n*ldb + cb + half*16    -> chunks at +0, +8  (B contiguous)
__device__ __forceinline__ v8f wmma_kloop(const unsigned short* __restrict__ arow,
                                          const unsigned short* __restrict__ brow,
                                          int K) {
  v8f acc = {};
  for (int k0 = 0; k0 < K; k0 += 32) {
    v16bf a = load_frag(arow + k0, 8);
    v16bf b = load_frag(brow + k0, 8);
    acc = __builtin_amdgcn_wmma_f32_16x16x32_bf16(false, a, false, b,
                                                  (short)0, acc, false, false);
  }
  return acc;
}

// ---------------------------------------------------------------------------
// 0) float -> bf16 conversion (weights)
// ---------------------------------------------------------------------------
__global__ void k_convert(const float* __restrict__ src,
                          unsigned short* __restrict__ dst, int n) {
  int i = blockIdx.x * 256 + threadIdx.x;
  if (i < n) dst[i] = f2bf(src[i]);
}

// ---------------------------------------------------------------------------
// 1) Trilinear gather:  Xg[n][k] = tri(base+off[k,0]) - tri(base+off[k,1])
// ---------------------------------------------------------------------------
__device__ __forceinline__ float tri_sample(const float* __restrict__ vol,
                                            float cx, float cy, float cz) {
  // ((c+1)*32 - 1) * 0.5 = 16*c + 15.5
  float ix = fmaf(cx, 16.f, 15.5f);
  float iy = fmaf(cy, 16.f, 15.5f);
  float iz = fmaf(cz, 16.f, 15.5f);
  float xf = floorf(ix), yf = floorf(iy), zf = floorf(iz);
  float fx = ix - xf, fy = iy - yf, fz = iz - zf;
  int x0 = (int)xf, y0 = (int)yf, z0 = (int)zf;
  float out = 0.f;
#pragma unroll
  for (int dz = 0; dz < 2; ++dz) {
    int zi = z0 + dz;
    float wz = dz ? fz : 1.f - fz;
    bool vz = (zi >= 0) && (zi < 32);
    int zc = zi < 0 ? 0 : (zi > 31 ? 31 : zi);
#pragma unroll
    for (int dy = 0; dy < 2; ++dy) {
      int yi = y0 + dy;
      float wy = dy ? fy : 1.f - fy;
      bool vy = (yi >= 0) && (yi < 32);
      int yc = yi < 0 ? 0 : (yi > 31 ? 31 : yi);
#pragma unroll
      for (int dx = 0; dx < 2; ++dx) {
        int xi = x0 + dx;
        float wx = dx ? fx : 1.f - fx;
        bool vx = (xi >= 0) && (xi < 32);
        int xc = xi < 0 ? 0 : (xi > 31 ? 31 : xi);
        float v = (vx && vy && vz) ? vol[(zc * 32 + yc) * 32 + xc] : 0.f;
        out = fmaf(wz * wy * wx, v, out);
      }
    }
  }
  return out;
}

__global__ void __launch_bounds__(256)
k_gather(const float* __restrict__ vol, const float* __restrict__ off,
         unsigned short* __restrict__ Xg) {
  __shared__ float soff[1024 * 6];
  int tid = threadIdx.x;
  for (int i = tid; i < 1024 * 6; i += 256) soff[i] = off[i];
  __syncthreads();
  int n = blockIdx.x * 256 + tid;
  int xv = n & 31, yv = (n >> 5) & 31, zv = (n >> 10) & 31;
  float bx = fmaf((float)xv, 0.0625f, 0.03125f) - 1.f;   // (2x+1)/32 - 1
  float by = fmaf((float)yv, 0.0625f, 0.03125f) - 1.f;
  float bz = fmaf((float)zv, 0.0625f, 0.03125f) - 1.f;
  unsigned short* dst = Xg + (size_t)n * 1024;
  for (int k = 0; k < 1024; ++k) {
    const float* o = &soff[k * 6];
    float s0 = tri_sample(vol, bx + o[0], by + o[1], bz + o[2]);
    float s1 = tri_sample(vol, bx + o[3], by + o[4], bz + o[5]);
    dst[k] = f2bf(s0 - s1);
  }
}

// ---------------------------------------------------------------------------
// 2) WMMA GEMM -> channel-major f32 Y.   A: (mtiles*16) x K bf16 row-major.
//    B: point-major bf16, row stride ldb. groupmode=1: cb = (mtile>>2)*256.
// ---------------------------------------------------------------------------
__global__ void __launch_bounds__(256)
k_gemm_cm(const unsigned short* __restrict__ A,
          const unsigned short* __restrict__ B, int ldb, int K, int groupmode,
          float* __restrict__ Y) {
  int lane = threadIdx.x & 31;
  int wid  = threadIdx.x >> 5;
  int gw = blockIdx.x * 8 + wid;
  int ntile = gw & (NTILES - 1);
  int mtile = gw >> 11;
  int m = lane & 15, half = lane >> 4;
  int cb = groupmode ? (mtile >> 2) * 256 : 0;
  const unsigned short* arow = A + (size_t)(mtile * 16 + m) * K + half * 16;
  const unsigned short* brow = B + (size_t)(ntile * 16 + m) * ldb + cb + half * 16;
  v8f acc = wmma_kloop(arow, brow, K);
  int n  = ntile * 16 + m;
  int mb = mtile * 16 + half * 8;
#pragma unroll
  for (int r = 0; r < 8; ++r)
    Y[(size_t)(mb + r) * NPTS + n] = acc[r];
}

// ---------------------------------------------------------------------------
// 3) BN stats over channel-major f32 Y, fold (g,b) -> (scale, shift)
// ---------------------------------------------------------------------------
__global__ void k_stats_cm(const float* __restrict__ Y, const float* __restrict__ g,
                           const float* __restrict__ b, float* __restrict__ scale,
                           float* __restrict__ shift) {
  int c = blockIdx.x, tid = threadIdx.x;
  __shared__ float s1[256], s2[256];
  const float* p = Y + (size_t)c * NPTS;
  float a = 0.f, q = 0.f;
  for (int i = tid; i < NPTS; i += 256) { float v = p[i]; a += v; q = fmaf(v, v, q); }
  s1[tid] = a; s2[tid] = q; __syncthreads();
  for (int s = 128; s > 0; s >>= 1) {
    if (tid < s) { s1[tid] += s1[tid + s]; s2[tid] += s2[tid + s]; }
    __syncthreads();
  }
  if (tid == 0) {
    float m = s1[0] * (1.f / NPTS);
    float var = s2[0] * (1.f / NPTS) - m * m;
    float inv = rsqrtf(var + 1e-5f);
    float sc = g[c] * inv;
    scale[c] = sc;
    shift[c] = b[c] - m * sc;
  }
}

// stats over point-major bf16 feature buffer (row length 256), channels cbase..
__global__ void k_stats_pm(const unsigned short* __restrict__ F, int cbase,
                           float* __restrict__ meanF, float* __restrict__ invF) {
  int c = cbase + blockIdx.x, tid = threadIdx.x;
  __shared__ float s1[256], s2[256];
  float a = 0.f, q = 0.f;
  for (int i = tid; i < NPTS; i += 256) {
    float v = bf2f(F[(size_t)i * 256 + c]);
    a += v; q = fmaf(v, v, q);
  }
  s1[tid] = a; s2[tid] = q; __syncthreads();
  for (int s = 128; s > 0; s >>= 1) {
    if (tid < s) { s1[tid] += s1[tid + s]; s2[tid] += s2[tid + s]; }
    __syncthreads();
  }
  if (tid == 0) {
    float m = s1[0] * (1.f / NPTS);
    float var = s2[0] * (1.f / NPTS) - m * m;
    meanF[c] = m;
    invF[c] = rsqrtf(var + 1e-5f);
  }
}

// ---------------------------------------------------------------------------
// 4) Apply BN affine (+optional relu), f32 channel-major -> bf16 point-major
// ---------------------------------------------------------------------------
__global__ void k_apply(const float* __restrict__ Y, const float* __restrict__ scale,
                        const float* __restrict__ shift,
                        unsigned short* __restrict__ dst, int relu) {
  int idx = blockIdx.x * 256 + threadIdx.x;   // idx = c*NPTS + n
  int c = idx >> 15, n = idx & (NPTS - 1);
  float v = fmaf(Y[idx], scale[c], shift[c]);
  if (relu) v = fmaxf(v, 0.f);
  dst[(size_t)n * 256 + c] = f2bf(v);
}

// ---------------------------------------------------------------------------
// 5) Fold BN affine into weights: W'[o][j] = w[o][j]*g[j]*inv[j] (bf16),
//    bias[o] = sum_j w[o][j]*(b[j]-mean[j]*g[j]*inv[j]) (+extra[o])
// ---------------------------------------------------------------------------
__global__ void k_fold(const float* __restrict__ w, int O, int J,
                       const float* __restrict__ g, const float* __restrict__ b,
                       const float* __restrict__ meanF, const float* __restrict__ invF,
                       const float* __restrict__ extra,
                       unsigned short* __restrict__ Wp, float* __restrict__ bias) {
  int o = blockIdx.x, j = threadIdx.x;
  __shared__ float red[256];
  float contrib = 0.f;
  if (j < J) {
    if (o < O) {
      float sc = g[j] * invF[j];
      float sh = b[j] - meanF[j] * sc;
      float wv = w[o * J + j];
      Wp[o * J + j] = f2bf(wv * sc);
      contrib = wv * sh;
    } else {
      Wp[o * J + j] = 0;   // zero-pad rows (for the 8->16 padded final layer)
    }
  }
  red[j] = contrib; __syncthreads();
  for (int s = 128; s > 0; s >>= 1) {
    if (j < s) red[j] += red[j + s];
    __syncthreads();
  }
  if (j == 0) bias[o] = red[0] + ((extra && o < O) ? extra[o] : 0.f);
}

// ---------------------------------------------------------------------------
// 6) Dense-block WMMA GEMM: acc = A * F[:, 0..K) ; relu(acc+bias) -> F col cbase_out
// ---------------------------------------------------------------------------
__global__ void __launch_bounds__(256)
k_gemm_f(const unsigned short* __restrict__ A, int K,
         const unsigned short* __restrict__ Fin,
         unsigned short* __restrict__ Fout, int cbase_out,
         const float* __restrict__ bias) {
  int lane = threadIdx.x & 31;
  int wid  = threadIdx.x >> 5;
  int gw = blockIdx.x * 8 + wid;
  int ntile = gw & (NTILES - 1);
  int mtile = gw >> 11;
  int m = lane & 15, half = lane >> 4;
  const unsigned short* arow = A + (size_t)(mtile * 16 + m) * K + half * 16;
  const unsigned short* brow = Fin + (size_t)(ntile * 16 + m) * 256 + half * 16;
  v8f acc = wmma_kloop(arow, brow, K);
  int n  = ntile * 16 + m;
  int ob = mtile * 16 + half * 8;
#pragma unroll
  for (int r = 0; r < 8; ++r) {
    float v = acc[r] + (bias ? bias[ob + r] : 0.f);
    v = fmaxf(v, 0.f);
    Fout[(size_t)n * 256 + cbase_out + ob + r] = f2bf(v);
  }
}

// ---------------------------------------------------------------------------
// 7) Output WMMA GEMM (A padded 16x256, valid rows 0..7) + sigmoid -> d_out
// ---------------------------------------------------------------------------
__global__ void __launch_bounds__(256)
k_gemm_out(const unsigned short* __restrict__ A,
           const unsigned short* __restrict__ F,
           const float* __restrict__ bias, float* __restrict__ out) {
  int lane = threadIdx.x & 31;
  int wid  = threadIdx.x >> 5;
  int ntile = blockIdx.x * 8 + wid;
  int m = lane & 15, half = lane >> 4;
  const unsigned short* arow = A + (size_t)m * 256 + half * 16;
  const unsigned short* brow = F + (size_t)(ntile * 16 + m) * 256 + half * 16;
  v8f acc = wmma_kloop(arow, brow, 256);
  if (half == 0) {                 // rows 0..7 are the L=8 real outputs
    int n = ntile * 16 + m;
#pragma unroll
    for (int r = 0; r < 8; ++r) {
      float v = acc[r] + bias[r];
      out[(size_t)r * NPTS + n] = 1.f / (1.f + __expf(-v));
    }
  }
}

// ---------------------------------------------------------------------------
extern "C" void kernel_launch(void* const* d_in, const int* in_sizes, int n_in,
                              void* d_out, int out_size, void* d_ws, size_t ws_size,
                              hipStream_t stream) {
  const float* images  = (const float*)d_in[0];   // (1,1,32,32,32)
  const float* offset1 = (const float*)d_in[1];   // (1024,2,3)
  const float* w1  = (const float*)d_in[2];       // (4,64,256)
  const float* g1  = (const float*)d_in[3];
  const float* b1  = (const float*)d_in[4];
  const float* w2  = (const float*)d_in[5];       // (128,256)
  const float* g2  = (const float*)d_in[6];
  const float* b2  = (const float*)d_in[7];
  const float* w3a = (const float*)d_in[8];       // (32,128)
  const float* w3b = (const float*)d_in[9];       // (32,160)
  const float* w3c = (const float*)d_in[10];      // (32,192)
  const float* w3d = (const float*)d_in[11];      // (32,224)
  const float* g3a = (const float*)d_in[12];
  const float* b3a = (const float*)d_in[13];
  const float* g3b = (const float*)d_in[14];
  const float* b3b = (const float*)d_in[15];
  const float* g3c = (const float*)d_in[16];
  const float* b3c = (const float*)d_in[17];
  const float* g3d = (const float*)d_in[18];
  const float* b3d = (const float*)d_in[19];
  const float* w4  = (const float*)d_in[20];      // (8,256)
  const float* b4  = (const float*)d_in[21];      // (8,)

  char* cur = (char*)d_ws;
  auto carve = [&](size_t bytes) -> char* {
    char* p = cur;
    cur += (bytes + 255) & ~(size_t)255;
    return p;
  };

  unsigned short* Xg  = (unsigned short*)carve((size_t)NPTS * 1024 * 2); // gathered feats
  float*          Yb  = (float*)carve((size_t)256 * NPTS * 4);           // gemm output (cm)
  unsigned short* X1b = (unsigned short*)carve((size_t)NPTS * 256 * 2);  // relu(bn1) feats
  unsigned short* Fb  = (unsigned short*)carve((size_t)NPTS * 256 * 2);  // dense-block feats
  unsigned short* WB1 = (unsigned short*)carve(4 * 64 * 256 * 2);
  unsigned short* WB2 = (unsigned short*)carve(128 * 256 * 2);
  unsigned short* WB3a= (unsigned short*)carve(32 * 128 * 2);
  unsigned short* W3bp= (unsigned short*)carve(32 * 160 * 2);
  unsigned short* W3cp= (unsigned short*)carve(32 * 192 * 2);
  unsigned short* W3dp= (unsigned short*)carve(32 * 224 * 2);
  unsigned short* W4p = (unsigned short*)carve(16 * 256 * 2);
  float* scale1 = (float*)carve(256 * 4);
  float* shift1 = (float*)carve(256 * 4);
  float* scale2 = (float*)carve(128 * 4);
  float* shift2 = (float*)carve(128 * 4);
  float* meanF  = (float*)carve(256 * 4);
  float* invF   = (float*)carve(256 * 4);
  float* bias3b = (float*)carve(32 * 4);
  float* bias3c = (float*)carve(32 * 4);
  float* bias3d = (float*)carve(32 * 4);
  float* bias4  = (float*)carve(16 * 4);

  // --- weight conversion (straight bf16; BN-folded layers converted later) ---
  k_convert<<<(4 * 64 * 256 + 255) / 256, 256, 0, stream>>>(w1, WB1, 4 * 64 * 256);
  k_convert<<<(128 * 256 + 255) / 256, 256, 0, stream>>>(w2, WB2, 128 * 256);
  k_convert<<<(32 * 128 + 255) / 256, 256, 0, stream>>>(w3a, WB3a, 32 * 128);

  // --- trilinear gather (volume L2-resident; offsets staged in LDS) ---
  k_gather<<<NPTS / 256, 256, 0, stream>>>(images, offset1, Xg);

  // --- grouped GEMM1: y1[256][N] ---
  k_gemm_cm<<<16 * NTILES / 8, 256, 0, stream>>>(WB1, Xg, 1024, 256, 1, Yb);
  k_stats_cm<<<256, 256, 0, stream>>>(Yb, g1, b1, scale1, shift1);
  k_apply<<<256 * NPTS / 256, 256, 0, stream>>>(Yb, scale1, shift1, X1b, 1);

  // --- GEMM2: y2[128][N], bn -> F channels 0..127 ---
  k_gemm_cm<<<8 * NTILES / 8, 256, 0, stream>>>(WB2, X1b, 256, 256, 0, Yb);
  k_stats_cm<<<128, 256, 0, stream>>>(Yb, g2, b2, scale2, shift2);
  k_apply<<<128 * NPTS / 256, 256, 0, stream>>>(Yb, scale2, shift2, Fb, 0);
  k_stats_pm<<<128, 256, 0, stream>>>(Fb, 0, meanF, invF);

  // --- dense block stage a: relu(w3a * F[0:128)) -> F[128:160) ---
  k_gemm_f<<<2 * NTILES / 8, 256, 0, stream>>>(WB3a, 128, Fb, Fb, 128, (const float*)nullptr);
  k_stats_pm<<<32, 256, 0, stream>>>(Fb, 128, meanF, invF);

  // --- stage b ---
  k_fold<<<32, 256, 0, stream>>>(w3b, 32, 160, g3a, b3a, meanF, invF,
                                 (const float*)nullptr, W3bp, bias3b);
  k_gemm_f<<<2 * NTILES / 8, 256, 0, stream>>>(W3bp, 160, Fb, Fb, 160, bias3b);
  k_stats_pm<<<32, 256, 0, stream>>>(Fb, 160, meanF, invF);

  // --- stage c ---
  k_fold<<<32, 256, 0, stream>>>(w3c, 32, 192, g3b, b3b, meanF, invF,
                                 (const float*)nullptr, W3cp, bias3c);
  k_gemm_f<<<2 * NTILES / 8, 256, 0, stream>>>(W3cp, 192, Fb, Fb, 192, bias3c);
  k_stats_pm<<<32, 256, 0, stream>>>(Fb, 192, meanF, invF);

  // --- stage d ---
  k_fold<<<32, 256, 0, stream>>>(w3d, 32, 224, g3c, b3c, meanF, invF,
                                 (const float*)nullptr, W3dp, bias3d);
  k_gemm_f<<<2 * NTILES / 8, 256, 0, stream>>>(W3dp, 224, Fb, Fb, 224, bias3d);
  k_stats_pm<<<32, 256, 0, stream>>>(Fb, 224, meanF, invF);

  // --- output layer: fold bn(g3d,b3d) + b4 into padded 16x256 weights ---
  k_fold<<<16, 256, 0, stream>>>(w4, 8, 256, g3d, b3d, meanF, invF, b4, W4p, bias4);
  k_gemm_out<<<NTILES / 8, 256, 0, stream>>>(W4p, Fb, bias4, (float*)d_out);
}
